// DeepseekV2MoE_52785148067908
// MI455X (gfx1250) — matrix operand order repaired and checked by reading, not compiled, not run on hardware
//
#include <hip/hip_runtime.h>

// ---------------- problem constants ----------------
static constexpr int T_  = 1024;
static constexpr int H_  = 2048;
static constexpr int I_  = 1408;
static constexpr int E_  = 16;
static constexpr int K_  = 6;
static constexpr int SI_ = I_ * 2;   // 2816

typedef __attribute__((ext_vector_type(16))) __bf16 v16bf;
typedef __attribute__((ext_vector_type(8)))  float  v8f;
typedef unsigned int uint_t;
typedef __attribute__((ext_vector_type(4)))  uint_t v4u;

union Frag { v4u u[2]; v16bf bf; };

// packed fp32->bf16 (RNE), gfx1250 VOP3
__device__ __forceinline__ uint_t f2bf2(float lo, float hi) {
    uint_t d;
    asm("v_cvt_pk_bf16_f32 %0, %1, %2" : "=v"(d) : "v"(lo), "v"(hi));
    return d;
}

// async global->LDS 16B copy per lane (ASYNCcnt-tracked DMA path)
__device__ __forceinline__ void async_b128(uint_t lds_addr, const void* gptr) {
    asm volatile("global_load_async_to_lds_b128 %0, %1, off"
                 :: "v"(lds_addr), "v"(gptr) : "memory");
}
__device__ __forceinline__ void wait_async0() {
    asm volatile("s_wait_asynccnt 0x0" ::: "memory");
}
__device__ __forceinline__ uint_t lds_addr_of(const void* p) {
    return (uint_t)(unsigned long long)p;   // low 32 bits = LDS offset
}

// A fragment (16x32 bf16): lanes0-15 K={0..15}, lanes16-31 K={16..31}
__device__ __forceinline__ v16bf load_fragA(const unsigned short* rowp, int kk, int lane) {
    const unsigned short* p = rowp + kk + ((lane >> 4) << 4);
    Frag f;
    f.u[0] = *(const v4u*)p;
    f.u[1] = *(const v4u*)(p + 8);
    return f.bf;
}
// B fragment (32x16 bf16): lane's column, 16 contiguous K, +16 for lanes>=16
__device__ __forceinline__ v16bf load_fragB(const unsigned short* rowp, int kk, int lane) {
    const unsigned short* p = rowp + kk + ((lane >> 4) << 4);
    Frag f;
    f.u[0] = *(const v4u*)p;
    f.u[1] = *(const v4u*)(p + 8);
    return f.bf;
}

#define WMMA_BF16(A, B, C) \
    __builtin_amdgcn_wmma_f32_16x16x32_bf16(false, (A), false, (B), (short)0, (C), false, false)

// ---------------------------------------------------------------------------
// Router: logits -> softmax -> top-6 -> renormalize -> dense comb[T,E]
// ---------------------------------------------------------------------------
__global__ __launch_bounds__(256) void moe_router(const float* __restrict__ x,
                                                  const float* __restrict__ gw,
                                                  float* __restrict__ comb) {
    int t = blockIdx.x * blockDim.x + threadIdx.x;
    if (t >= T_) return;
    float logits[E_];
#pragma unroll
    for (int e = 0; e < E_; ++e) logits[e] = 0.f;
    const float* xr = x + (size_t)t * H_;
    for (int h = 0; h < H_; ++h) {
        float xv = xr[h];
#pragma unroll
        for (int e = 0; e < E_; ++e) logits[e] += xv * gw[e * H_ + h];
    }
    float mx = logits[0];
#pragma unroll
    for (int e = 1; e < E_; ++e) mx = fmaxf(mx, logits[e]);
    float p[E_]; float sum = 0.f;
#pragma unroll
    for (int e = 0; e < E_; ++e) { p[e] = __expf(logits[e] - mx); sum += p[e]; }
    float inv = 1.f / sum;
#pragma unroll
    for (int e = 0; e < E_; ++e) p[e] *= inv;

    float cw[E_]; bool used[E_];
#pragma unroll
    for (int e = 0; e < E_; ++e) { cw[e] = 0.f; used[e] = false; }
    float wsum = 0.f;
#pragma unroll
    for (int k = 0; k < K_; ++k) {
        int best = 0; float bv = -1.f;
#pragma unroll
        for (int e = 0; e < E_; ++e)
            if (!used[e] && p[e] > bv) { bv = p[e]; best = e; }
        used[best] = true; cw[best] = bv; wsum += bv;
    }
    float winv = 1.f / wsum;
#pragma unroll
    for (int e = 0; e < E_; ++e) comb[t * E_ + e] = cw[e] * winv;
}

// ---------------------------------------------------------------------------
// x fp32 -> bf16 once (reused by 16 expert tiles + shared + removes cvt work)
// ---------------------------------------------------------------------------
__global__ __launch_bounds__(256) void convert_x(const float* __restrict__ x,
                                                 unsigned short* __restrict__ xb) {
    int idx = blockIdx.x * 256 + threadIdx.x;     // 8 floats per thread
    const float4* src = (const float4*)x + (size_t)idx * 2;
    float4 a = src[0], b = src[1];
    v4u o;
    o.x = f2bf2(a.x, a.y); o.y = f2bf2(a.z, a.w);
    o.z = f2bf2(b.x, b.y); o.w = f2bf2(b.z, b.w);
    ((v4u*)xb)[idx] = o;
}

// ---------------------------------------------------------------------------
// Fused gate+up GEMM + SwiGLU + combine epilogue -> bf16 h
//   block tile 128(tok) x 64(i), KC=32; A tile via async DMA, B via cvt+ds
//   8 waves: 4 row-tiles(32) x 2 col-tiles(32); 8 wmma per chunk per wave
// ---------------------------------------------------------------------------
#define MKC   32
#define MKCP  40   // padded row (shorts) = 80B (16B-aligned)
__global__ __launch_bounds__(256)
void moe_mlp_in(const unsigned short* __restrict__ xb,  // [T,H] bf16
                const float* __restrict__ wg_all,
                const float* __restrict__ wu_all,
                const float* __restrict__ comb,         // null => weight 1
                unsigned short* __restrict__ hb,
                int Idim, int ldt)
{
    __shared__ unsigned short As[2][128][MKCP];
    __shared__ unsigned short Bg[2][64][MKCP];
    __shared__ unsigned short Bu[2][64][MKCP];

    const int e  = blockIdx.z;
    const int i0 = blockIdx.x * 64;
    const int t0 = blockIdx.y * 128;
    const float* wg = wg_all + (size_t)e * Idim * H_ + (size_t)i0 * H_;
    const float* wu = wu_all + (size_t)e * Idim * H_ + (size_t)i0 * H_;
    const unsigned short* xa = xb + (size_t)t0 * H_;
    unsigned short* hbp = hb + (size_t)e * Idim;

    const int tid  = threadIdx.x;
    const int lane = tid & 31;
    const int wid  = tid >> 5;
    const int wm   = wid & 3;
    const int wn   = wid >> 2;

    // A staging: 128 rows x 4 chunks(16B); 2 chunks/thread
    const int ar = tid >> 1, ac = tid & 1;
    // B staging: 64 rows x 8 float4; 2 float4/thread per matrix
    const int br = tid >> 2, bc = tid & 3;

    float4 pg[2], pu[2];

    auto asyncA = [&](int k0, int b) {
#pragma unroll
        for (int j = 0; j < 2; ++j) {
            int ch = ac + 2 * j;                 // 16B chunk 0..3
            async_b128(lds_addr_of(&As[b][ar][ch * 8]),
                       xa + (size_t)ar * H_ + k0 + ch * 8);
        }
    };
    auto gloadB = [&](int k0) {
        const float4* gr = (const float4*)(wg + (size_t)br * H_ + k0);
        const float4* ur = (const float4*)(wu + (size_t)br * H_ + k0);
#pragma unroll
        for (int j = 0; j < 2; ++j) { pg[j] = gr[bc + 4 * j]; pu[j] = ur[bc + 4 * j]; }
    };
    auto lstoreB = [&](int b) {
#pragma unroll
        for (int j = 0; j < 2; ++j) {
            int c = bc + 4 * j;
            uint2 v;
            v.x = f2bf2(pg[j].x, pg[j].y); v.y = f2bf2(pg[j].z, pg[j].w);
            *(uint2*)&Bg[b][br][c * 4] = v;
            v.x = f2bf2(pu[j].x, pu[j].y); v.y = f2bf2(pu[j].z, pu[j].w);
            *(uint2*)&Bu[b][br][c * 4] = v;
        }
    };

    v8f ag[2][2], au[2][2];
#pragma unroll
    for (int mi = 0; mi < 2; ++mi)
#pragma unroll
        for (int ni = 0; ni < 2; ++ni) { ag[mi][ni] = (v8f)(0.f); au[mi][ni] = (v8f)(0.f); }

    asyncA(0, 0);
    gloadB(0);
    int buf = 0;
    const int nch = H_ / MKC;
    for (int c = 0; c < nch; ++c) {
        lstoreB(buf);
        wait_async0();            // A chunk c landed in LDS[buf]
        __syncthreads();
        if (c + 1 < nch) { asyncA((c + 1) * MKC, buf ^ 1); gloadB((c + 1) * MKC); }
        v16bf a0 = load_fragA(&As[buf][wm * 32 +      (lane & 15)][0], 0, lane);
        v16bf a1 = load_fragA(&As[buf][wm * 32 + 16 + (lane & 15)][0], 0, lane);
        v16bf g0 = load_fragB(&Bg[buf][wn * 32 +      (lane & 15)][0], 0, lane);
        v16bf g1 = load_fragB(&Bg[buf][wn * 32 + 16 + (lane & 15)][0], 0, lane);
        v16bf u0 = load_fragB(&Bu[buf][wn * 32 +      (lane & 15)][0], 0, lane);
        v16bf u1 = load_fragB(&Bu[buf][wn * 32 + 16 + (lane & 15)][0], 0, lane);
        ag[0][0] = WMMA_BF16(a0, g0, ag[0][0]);
        ag[0][1] = WMMA_BF16(a0, g1, ag[0][1]);
        ag[1][0] = WMMA_BF16(a1, g0, ag[1][0]);
        ag[1][1] = WMMA_BF16(a1, g1, ag[1][1]);
        au[0][0] = WMMA_BF16(a0, u0, au[0][0]);
        au[0][1] = WMMA_BF16(a0, u1, au[0][1]);
        au[1][0] = WMMA_BF16(a1, u0, au[1][0]);
        au[1][1] = WMMA_BF16(a1, u1, au[1][1]);
        buf ^= 1;
    }

    // epilogue: C/D row = r + 8*(lane>=16), col = lane&15
#pragma unroll
    for (int mi = 0; mi < 2; ++mi) {
        const int mbase = t0 + wm * 32 + mi * 16 + ((lane >> 4) << 3);
        const int nbase = i0 + wn * 32 + (lane & 15);
#pragma unroll
        for (int r = 0; r < 8; ++r) {
            int t = mbase + r;
            float w = comb ? comb[t * E_ + e] : 1.0f;
            float g0v = ag[mi][0][r], u0v = au[mi][0][r];
            float g1v = ag[mi][1][r], u1v = au[mi][1][r];
            float h0 = w * g0v * __builtin_amdgcn_rcpf(1.f + __expf(-g0v)) * u0v;
            float h1 = w * g1v * __builtin_amdgcn_rcpf(1.f + __expf(-g1v)) * u1v;
            uint_t pk = f2bf2(h0, h1);
            hbp[(size_t)t * ldt + nbase]      = (unsigned short)(pk & 0xffffu);
            hbp[(size_t)t * ldt + nbase + 16] = (unsigned short)(pk >> 16);
        }
    }
}

// ---------------------------------------------------------------------------
// Down projection: out[t,n] (+)= sum_k hb[t,k] * W[e(k), n, i(k)]
//   block tile 128(tok) x 64(h), KC=64; A via async DMA, B via cvt+ds
// ---------------------------------------------------------------------------
#define DKC   64
#define DKCP  72
__global__ __launch_bounds__(256)
void moe_down(const unsigned short* __restrict__ hb,   // [T, Ktot] bf16
              const float* __restrict__ wdown,
              float* __restrict__ out,
              int Ktot, int Iper, long estride, int ldn, int accumulate)
{
    __shared__ unsigned short As[2][128][DKCP];
    __shared__ unsigned short Bs[2][64][DKCP];

    const int h0 = blockIdx.x * 64;
    const int t0 = blockIdx.y * 128;

    const int tid  = threadIdx.x;
    const int lane = tid & 31;
    const int wid  = tid >> 5;
    const int wm   = wid & 3;
    const int wn   = wid >> 2;

    const int ar = tid >> 1, ac = tid & 1;   // A: 4x16B chunks of 8/row
    const int br = tid >> 2, bc = tid & 3;   // B: 4 float4 of 16/row

    float4 pb[4];

    auto asyncA = [&](int k0, int b) {
        const unsigned short* src = hb + (size_t)(t0 + ar) * Ktot + k0;
#pragma unroll
        for (int j = 0; j < 4; ++j) {
            int ch = ac + 2 * j;                 // 16B chunk 0..7
            async_b128(lds_addr_of(&As[b][ar][ch * 8]), src + ch * 8);
        }
    };
    auto gloadB = [&](int k0) {
        const int e  = k0 / Iper;                // DKC | Iper
        const int ik = k0 - e * Iper;
        const float4* brow = (const float4*)(wdown + (size_t)e * estride +
                                             (size_t)(h0 + br) * ldn + ik);
#pragma unroll
        for (int j = 0; j < 4; ++j) pb[j] = brow[bc + 4 * j];
    };
    auto lstoreB = [&](int b) {
#pragma unroll
        for (int j = 0; j < 4; ++j) {
            int c = bc + 4 * j;
            uint2 v; v.x = f2bf2(pb[j].x, pb[j].y); v.y = f2bf2(pb[j].z, pb[j].w);
            *(uint2*)&Bs[b][br][c * 4] = v;
        }
    };

    v8f acc[2][2];
#pragma unroll
    for (int mi = 0; mi < 2; ++mi)
#pragma unroll
        for (int ni = 0; ni < 2; ++ni) acc[mi][ni] = (v8f)(0.f);

    asyncA(0, 0);
    gloadB(0);
    int buf = 0;
    const int nch = Ktot / DKC;
    for (int c = 0; c < nch; ++c) {
        lstoreB(buf);
        wait_async0();
        __syncthreads();
        if (c + 1 < nch) { asyncA((c + 1) * DKC, buf ^ 1); gloadB((c + 1) * DKC); }
#pragma unroll
        for (int kk = 0; kk < DKC; kk += 32) {
            v16bf a0 = load_fragA(&As[buf][wm * 32 +      (lane & 15)][0], kk, lane);
            v16bf a1 = load_fragA(&As[buf][wm * 32 + 16 + (lane & 15)][0], kk, lane);
            v16bf b0 = load_fragB(&Bs[buf][wn * 32 +      (lane & 15)][0], kk, lane);
            v16bf b1 = load_fragB(&Bs[buf][wn * 32 + 16 + (lane & 15)][0], kk, lane);
            acc[0][0] = WMMA_BF16(a0, b0, acc[0][0]);
            acc[0][1] = WMMA_BF16(a0, b1, acc[0][1]);
            acc[1][0] = WMMA_BF16(a1, b0, acc[1][0]);
            acc[1][1] = WMMA_BF16(a1, b1, acc[1][1]);
        }
        buf ^= 1;
    }

#pragma unroll
    for (int mi = 0; mi < 2; ++mi) {
        const int mbase = t0 + wm * 32 + mi * 16 + ((lane >> 4) << 3);
        const int nbase = h0 + wn * 32 + (lane & 15);
#pragma unroll
        for (int r = 0; r < 8; ++r) {
            size_t idx = (size_t)(mbase + r) * H_ + nbase;
            if (accumulate) {
                out[idx]      += acc[mi][0][r];
                out[idx + 16] += acc[mi][1][r];
            } else {
                out[idx]      = acc[mi][0][r];
                out[idx + 16] = acc[mi][1][r];
            }
        }
    }
}

// ---------------------------------------------------------------------------
extern "C" void kernel_launch(void* const* d_in, const int* in_sizes, int n_in,
                              void* d_out, int out_size, void* d_ws, size_t ws_size,
                              hipStream_t stream) {
    const float* x       = (const float*)d_in[0];
    const float* gate_w  = (const float*)d_in[1];
    const float* w_gate  = (const float*)d_in[2];
    const float* w_up    = (const float*)d_in[3];
    const float* w_down  = (const float*)d_in[4];
    const float* sw_gate = (const float*)d_in[5];
    const float* sw_up   = (const float*)d_in[6];
    const float* sw_down = (const float*)d_in[7];
    float* out = (float*)d_out;

    char* ws = (char*)d_ws;
    float* comb = (float*)ws;                                   // T*E fp32
    size_t off = ((size_t)T_ * E_ * sizeof(float) + 255) & ~(size_t)255;
    unsigned short* xb = (unsigned short*)(ws + off);           // [T,H] bf16
    unsigned short* hb = xb + (size_t)T_ * H_;                  // [T, E*I] bf16
    unsigned short* hs = hb + (size_t)T_ * E_ * I_;             // [T, SI] bf16

    // 1) router -> comb ; x -> bf16
    moe_router<<<T_ / 256, 256, 0, stream>>>(x, gate_w, comb);
    convert_x<<<(T_ * H_ / 8) / 256, 256, 0, stream>>>(x, xb);

    // 2) routed experts: h = comb * silu(x@wg^T) * (x@wu^T)  (bf16)
    dim3 g_rt(I_ / 64, T_ / 128, E_);
    moe_mlp_in<<<g_rt, 256, 0, stream>>>(xb, w_gate, w_up, comb, hb, I_, E_ * I_);

    // 3) shared expert: hs = silu(x@swg^T) * (x@swu^T)
    dim3 g_sh(SI_ / 64, T_ / 128, 1);
    moe_mlp_in<<<g_sh, 256, 0, stream>>>(xb, sw_gate, sw_up, nullptr, hs, SI_, SI_);

    // 4) routed down-projection (initializes out)
    dim3 g_dn(H_ / 64, T_ / 128, 1);
    moe_down<<<g_dn, 256, 0, stream>>>(hb, w_down, out,
                                       E_ * I_, I_, (long)H_ * I_, I_, 0);

    // 5) shared down-projection (accumulates into out)
    moe_down<<<g_dn, 256, 0, stream>>>(hs, sw_down, out,
                                       SI_, SI_, 0L, SI_, 1);
}
